// GATv2_NeighSampler_68247030333462
// MI455X (gfx1250) — matrix operand order repaired and checked by reading, not compiled, not run on hardware
//
#include <hip/hip_runtime.h>
#include <hip/hip_bf16.h>

// ---------------- problem constants ----------------
#define N1V 160000
#define N2V 40000
#define N3V 10000
#define E1V 800000
#define E2V 200000
#define INV 128
#define HIDV 64
#define OUTV 64
#define H1V 4
#define H2V 1
#define D1V (H1V * HIDV)   // 256
#define NEG_SLOPE 0.2f
#define BN_EPS 1e-5f
#define SOFTMAX_EPS 1e-16f

typedef __attribute__((ext_vector_type(16))) __bf16 v16bf;
typedef __attribute__((ext_vector_type(8)))  __bf16 v8bf;
typedef __attribute__((ext_vector_type(8)))  float  v8f;

// ---------------- bf16 helpers (hardware cvt via C casts) ----------------
__device__ __forceinline__ float  bf2f(__bf16 h) { return (float)h; }
__device__ __forceinline__ __bf16 f2bf(float f)  { return (__bf16)f; }

// float atomic max via int ordering trick (works with -inf init)
__device__ __forceinline__ void atomicMaxF(float* addr, float val) {
    if (val >= 0.0f) {
        atomicMax((int*)addr, __float_as_int(val));
    } else {
        atomicMin((unsigned int*)addr, (unsigned int)__float_as_int(val));
    }
}

// ---------------- generic fill ----------------
__global__ void fill_f32_kernel(float* __restrict__ p, float v, size_t n) {
    size_t i = (size_t)blockIdx.x * blockDim.x + threadIdx.x;
    if (i < n) p[i] = v;
}

// ---------------- f32 -> bf16 convert ----------------
__global__ void conv_bf16_kernel(const float* __restrict__ in, __bf16* __restrict__ out, size_t n) {
    size_t i = (size_t)blockIdx.x * blockDim.x + threadIdx.x;
    if (i < n) out[i] = f2bf(in[i]);
}

// ---------------- f32 W[K,N] -> bf16 Wt[N,K] transpose-convert (weights are tiny) ----------------
__global__ void conv_transpose_kernel(const float* __restrict__ W, __bf16* __restrict__ Wt,
                                      int K, int N) {
    int i = blockIdx.x * blockDim.x + threadIdx.x;
    if (i >= K * N) return;
    int k = i / N, n = i % N;
    Wt[(size_t)n * K + k] = f2bf(W[i]);
}

// ---------------- WMMA bf16 GEMM ----------------
// Out[M,N](bf16) = A[M,K](bf16, row-major) @ Bt[N,K](bf16, i.e. B transposed) + bias(f32)
// One wave computes 4 consecutive 16x16 M-tiles of one N-tile (64x16 of output),
// reusing each B fragment across 4 back-to-back WMMAs. Fragments per ISA 7.12.2:
//   A 16x32 : lanes 0-15 -> K {kb..kb+7, kb+16..kb+23} with kb=k0 (lanes<16) / k0+8
//   B 32x16 : lane col = n0+(lane&15), K {kb..kb+7, kb+16..kb+23} with kb=k0 (lanes<16) / k0+8
// M-tail handling: tile row bases are CLAMPED to M-16 before the k-loop, so the
// inner loop is completely branch-free (all loads valid, 4 unconditional WMMAs,
// EXEC all-ones throughout). Only epilogue stores are guarded.
__global__ void wmma_gemm_bias_kernel(const __bf16* __restrict__ A, const __bf16* __restrict__ Bt,
                                      const float* __restrict__ bias, __bf16* __restrict__ Out,
                                      int M, int N, int K) {
    int wave = (int)(((size_t)blockIdx.x * blockDim.x + threadIdx.x) >> 5);
    int lane = threadIdx.x & 31;
    int tilesN = N >> 4;
    int m0 = (wave / tilesN) << 6;          // 64 rows per wave
    int n0 = (wave % tilesN) << 4;
    if (m0 >= M) return;                    // wave-uniform exit

    int half = lane >> 4;                   // 0: lanes 0-15, 1: lanes 16-31
    int l15  = lane & 15;
    int bcol = n0 + l15;

    // clamped tile row bases (M always multiple of 16)
    int r0 = m0;
    int r1 = (m0 + 16 < M) ? m0 + 16 : M - 16;
    int r2 = (m0 + 32 < M) ? m0 + 32 : M - 16;
    int r3 = (m0 + 48 < M) ? m0 + 48 : M - 16;

    const __bf16* Bp = Bt + (size_t)bcol * K + (half << 3);
    const __bf16* A0 = A + (size_t)(r0 + l15) * K + (half << 3);
    const __bf16* A1 = A + (size_t)(r1 + l15) * K + (half << 3);
    const __bf16* A2 = A + (size_t)(r2 + l15) * K + (half << 3);
    const __bf16* A3 = A + (size_t)(r3 + l15) * K + (half << 3);

    v8f acc0 = (v8f){0.f,0.f,0.f,0.f,0.f,0.f,0.f,0.f};
    v8f acc1 = acc0, acc2 = acc0, acc3 = acc0;

    for (int k0 = 0; k0 < K; k0 += 32) {
        v8bf b0 = *(const v8bf*)(Bp + k0);
        v8bf b1 = *(const v8bf*)(Bp + k0 + 16);
        v16bf bfrag = __builtin_shufflevector(b0, b1, 0,1,2,3,4,5,6,7,8,9,10,11,12,13,14,15);

        v8bf a0, a1;
        v16bf afrag;
        a0 = *(const v8bf*)(A0 + k0); a1 = *(const v8bf*)(A0 + k0 + 16);
        afrag = __builtin_shufflevector(a0, a1, 0,1,2,3,4,5,6,7,8,9,10,11,12,13,14,15);
        acc0 = __builtin_amdgcn_wmma_f32_16x16x32_bf16(false, afrag, false, bfrag, (short)0, acc0, false, false);

        a0 = *(const v8bf*)(A1 + k0); a1 = *(const v8bf*)(A1 + k0 + 16);
        afrag = __builtin_shufflevector(a0, a1, 0,1,2,3,4,5,6,7,8,9,10,11,12,13,14,15);
        acc1 = __builtin_amdgcn_wmma_f32_16x16x32_bf16(false, afrag, false, bfrag, (short)0, acc1, false, false);

        a0 = *(const v8bf*)(A2 + k0); a1 = *(const v8bf*)(A2 + k0 + 16);
        afrag = __builtin_shufflevector(a0, a1, 0,1,2,3,4,5,6,7,8,9,10,11,12,13,14,15);
        acc2 = __builtin_amdgcn_wmma_f32_16x16x32_bf16(false, afrag, false, bfrag, (short)0, acc2, false, false);

        a0 = *(const v8bf*)(A3 + k0); a1 = *(const v8bf*)(A3 + k0 + 16);
        afrag = __builtin_shufflevector(a0, a1, 0,1,2,3,4,5,6,7,8,9,10,11,12,13,14,15);
        acc3 = __builtin_amdgcn_wmma_f32_16x16x32_bf16(false, afrag, false, bfrag, (short)0, acc3, false, false);
    }

    float bv = bias[bcol];
    // D layout: VGPR r -> row base + r (lanes 0-15) / base + 8 + r (lanes 16-31)
    int rofs = (half << 3);
    #pragma unroll
    for (int r = 0; r < 8; ++r)
        Out[(size_t)(r0 + rofs + r) * N + bcol] = f2bf(acc0[r] + bv);
    if (m0 + 16 < M) {
        #pragma unroll
        for (int r = 0; r < 8; ++r)
            Out[(size_t)(r1 + rofs + r) * N + bcol] = f2bf(acc1[r] + bv);
    }
    if (m0 + 32 < M) {
        #pragma unroll
        for (int r = 0; r < 8; ++r)
            Out[(size_t)(r2 + rofs + r) * N + bcol] = f2bf(acc2[r] + bv);
    }
    if (m0 + 48 < M) {
        #pragma unroll
        for (int r = 0; r < 8; ++r)
            Out[(size_t)(r3 + rofs + r) * N + bcol] = f2bf(acc3[r] + bv);
    }
}

// ---------------- edge attention logits + segment max ----------------
template <int H>
__global__ void edge_alpha_kernel(const __bf16* __restrict__ xl, const __bf16* __restrict__ xr,
                                  const int* __restrict__ src, const int* __restrict__ dst,
                                  const float* __restrict__ att,
                                  float* __restrict__ alpha, float* __restrict__ mmax, int E) {
    int e = (int)(((size_t)blockIdx.x * blockDim.x + threadIdx.x) >> 5);
    int lane = threadIdx.x & 31;
    if (e >= E) return;
    int s = src[e], d = dst[e];
    const __bf16* xs = xl + (size_t)s * (H * 64);
    const __bf16* xd = xr + (size_t)d * (H * 64);
    #pragma unroll
    for (int h = 0; h < H; ++h) {
        float p = 0.f;
        #pragma unroll
        for (int i = 0; i < 2; ++i) {
            int c = lane + i * 32;
            float v = bf2f(xs[h * 64 + c]) + bf2f(xd[h * 64 + c]);
            v = v > 0.f ? v : NEG_SLOPE * v;
            p += att[h * 64 + c] * v;
        }
        #pragma unroll
        for (int off = 16; off > 0; off >>= 1) p += __shfl_xor(p, off, 32);
        if (lane == 0) {
            alpha[(size_t)e * H + h] = p;
            atomicMaxF(&mmax[(size_t)d * H + h], p);
        }
    }
}

// ---------------- exp(alpha - max) + segment sum ----------------
template <int H>
__global__ void edge_exp_kernel(float* __restrict__ alpha, const float* __restrict__ mmax,
                                float* __restrict__ denom, const int* __restrict__ dst, int E) {
    size_t i = (size_t)blockIdx.x * blockDim.x + threadIdx.x;
    if (i >= (size_t)E * H) return;
    int e = (int)(i / H), h = (int)(i % H);
    int d = dst[e];
    float a = __expf(alpha[i] - mmax[(size_t)d * H + h]);
    alpha[i] = a;
    atomicAdd(&denom[(size_t)d * H + h], a);
}

// ---------------- weighted scatter-add aggregation ----------------
template <int H>
__global__ void edge_aggr_kernel(const __bf16* __restrict__ xl, const float* __restrict__ a,
                                 const float* __restrict__ denom,
                                 const int* __restrict__ src, const int* __restrict__ dst,
                                 float* __restrict__ out, int E) {
    int e = blockIdx.x;
    int c = threadIdx.x;     // 0 .. H*64-1
    int h = c >> 6;
    int s = src[e], d = dst[e];
    float coef = a[(size_t)e * H + h] / (denom[(size_t)d * H + h] + SOFTMAX_EPS);
    atomicAdd(&out[(size_t)d * (H * 64) + c], coef * bf2f(xl[(size_t)s * (H * 64) + c]));
}

// ---------------- batchnorm stats (sum / sumsq per channel) ----------------
__global__ void bn_stats_kernel(const float* __restrict__ h, const float* __restrict__ bias1,
                                float* __restrict__ sum, float* __restrict__ sumsq, int rows) {
    int c = threadIdx.x;               // 256 channels
    int r0 = blockIdx.x * 64;
    float b = bias1[c];
    float s = 0.f, sq = 0.f;
    for (int r = r0; r < r0 + 64 && r < rows; ++r) {
        float v = h[(size_t)r * D1V + c] + b;
        s += v; sq += v * v;
    }
    atomicAdd(&sum[c], s);
    atomicAdd(&sumsq[c], sq);
}

// ---------------- batchnorm apply + relu -> bf16 ----------------
__global__ void bn_apply_kernel(const float* __restrict__ h, const float* __restrict__ bias1,
                                const float* __restrict__ sum, const float* __restrict__ sumsq,
                                const float* __restrict__ gamma, const float* __restrict__ beta,
                                __bf16* __restrict__ out, int rows) {
    int c = threadIdx.x;               // 256
    int r = blockIdx.x;
    float inv = 1.0f / (float)rows;
    float mu  = sum[c] * inv;
    float var = sumsq[c] * inv - mu * mu;
    float v = h[(size_t)r * D1V + c] + bias1[c];
    v = gamma[c] * (v - mu) * rsqrtf(var + BN_EPS) + beta[c];
    v = v > 0.f ? v : 0.f;
    out[(size_t)r * D1V + c] = f2bf(v);
}

// ---------------- bias + log_softmax over 64 channels ----------------
__global__ void logsoftmax_kernel(const float* __restrict__ x, const float* __restrict__ bias,
                                  float* __restrict__ out, int rows) {
    int r = (int)(((size_t)blockIdx.x * blockDim.x + threadIdx.x) >> 5);
    int lane = threadIdx.x & 31;
    if (r >= rows) return;
    float v0 = x[(size_t)r * 64 + lane]      + bias[lane];
    float v1 = x[(size_t)r * 64 + lane + 32] + bias[lane + 32];
    float m = fmaxf(v0, v1);
    #pragma unroll
    for (int off = 16; off > 0; off >>= 1) m = fmaxf(m, __shfl_xor(m, off, 32));
    float s = __expf(v0 - m) + __expf(v1 - m);
    #pragma unroll
    for (int off = 16; off > 0; off >>= 1) s += __shfl_xor(s, off, 32);
    float ls = __logf(s);
    out[(size_t)r * 64 + lane]      = v0 - m - ls;
    out[(size_t)r * 64 + lane + 32] = v1 - m - ls;
}

// ---------------- launcher ----------------
extern "C" void kernel_launch(void* const* d_in, const int* in_sizes, int n_in,
                              void* d_out, int out_size, void* d_ws, size_t ws_size,
                              hipStream_t stream) {
    const float* x       = (const float*)d_in[0];
    const int*   ei1_src = (const int*)d_in[1];
    const int*   ei1_dst = (const int*)d_in[2];
    const int*   ei2_src = (const int*)d_in[3];
    const int*   ei2_dst = (const int*)d_in[4];
    const float* W1l     = (const float*)d_in[5];
    const float* b1l     = (const float*)d_in[6];
    const float* W1r     = (const float*)d_in[7];
    const float* b1r     = (const float*)d_in[8];
    const float* att1    = (const float*)d_in[9];
    const float* bias1   = (const float*)d_in[10];
    const float* gamma   = (const float*)d_in[11];
    const float* beta    = (const float*)d_in[12];
    const float* W2l     = (const float*)d_in[13];
    const float* b2l     = (const float*)d_in[14];
    const float* W2r     = (const float*)d_in[15];
    const float* b2r     = (const float*)d_in[16];
    const float* att2    = (const float*)d_in[17];
    const float* bias2   = (const float*)d_in[18];
    float* out           = (float*)d_out;

    // workspace layout (all 256B aligned -> vector loads safe)
    size_t off = 0;
    auto alloc = [&](size_t bytes) {
        void* p = (char*)d_ws + off;
        off += (bytes + 255) & ~(size_t)255;
        return p;
    };
    __bf16* x_bf   = (__bf16*)alloc((size_t)N1V * INV * 2);
    __bf16* W1lt   = (__bf16*)alloc((size_t)INV * D1V * 2);
    __bf16* W1rt   = (__bf16*)alloc((size_t)INV * D1V * 2);
    __bf16* W2lt   = (__bf16*)alloc((size_t)D1V * OUTV * 2);
    __bf16* W2rt   = (__bf16*)alloc((size_t)D1V * OUTV * 2);
    __bf16* xl_bf  = (__bf16*)alloc((size_t)N1V * D1V * 2);
    __bf16* xr_bf  = (__bf16*)alloc((size_t)N2V * D1V * 2);
    float*  alpha1 = (float*) alloc((size_t)E1V * H1V * 4);
    float*  m1     = (float*) alloc((size_t)N2V * H1V * 4);
    float*  den1   = (float*) alloc((size_t)N2V * H1V * 4);
    float*  hacc   = (float*) alloc((size_t)N2V * D1V * 4);
    float*  bsum   = (float*) alloc((size_t)D1V * 4);
    float*  bsq    = (float*) alloc((size_t)D1V * 4);
    __bf16* hbn_bf = (__bf16*)alloc((size_t)N2V * D1V * 2);
    __bf16* yl_bf  = (__bf16*)alloc((size_t)N2V * OUTV * 2);
    __bf16* yr_bf  = (__bf16*)alloc((size_t)N3V * OUTV * 2);
    float*  alpha2 = (float*) alloc((size_t)E2V * H2V * 4);
    float*  m2     = (float*) alloc((size_t)N3V * H2V * 4);
    float*  den2   = (float*) alloc((size_t)N3V * H2V * 4);
    float*  out2   = (float*) alloc((size_t)N3V * OUTV * 4);
    (void)ws_size;

    const float NEG_INF = -__builtin_inff();
    auto fill = [&](float* p, float v, size_t n) {
        fill_f32_kernel<<<(unsigned)((n + 255) / 256), 256, 0, stream>>>(p, v, n);
    };
    // waves = ceil(M/64) * (N/16); 8 waves per 256-thread block
    auto gemm_blocks = [](int M, int N) {
        size_t waves = (size_t)((M + 63) / 64) * (N / 16);
        return (unsigned)((waves + 7) / 8);
    };

    // ---- precision staging: bf16 activations + transposed bf16 weights ----
    conv_bf16_kernel<<<(unsigned)(((size_t)N1V * INV + 255) / 256), 256, 0, stream>>>(x, x_bf, (size_t)N1V * INV);
    conv_transpose_kernel<<<(INV * D1V + 255) / 256, 256, 0, stream>>>(W1l, W1lt, INV, D1V);
    conv_transpose_kernel<<<(INV * D1V + 255) / 256, 256, 0, stream>>>(W1r, W1rt, INV, D1V);
    conv_transpose_kernel<<<(D1V * OUTV + 255) / 256, 256, 0, stream>>>(W2l, W2lt, D1V, OUTV);
    conv_transpose_kernel<<<(D1V * OUTV + 255) / 256, 256, 0, stream>>>(W2r, W2rt, D1V, OUTV);

    // ---- layer 1 projections (bf16 WMMA) ----
    wmma_gemm_bias_kernel<<<gemm_blocks(N1V, D1V), 256, 0, stream>>>(x_bf, W1lt, b1l, xl_bf, N1V, D1V, INV);
    wmma_gemm_bias_kernel<<<gemm_blocks(N2V, D1V), 256, 0, stream>>>(x_bf, W1rt, b1r, xr_bf, N2V, D1V, INV);

    // ---- layer 1 segment softmax + aggregation ----
    fill(m1, NEG_INF, (size_t)N2V * H1V);
    fill(den1, 0.f, (size_t)N2V * H1V);
    fill(hacc, 0.f, (size_t)N2V * D1V);
    fill(bsum, 0.f, D1V);
    fill(bsq, 0.f, D1V);
    edge_alpha_kernel<H1V><<<(E1V + 7) / 8, 256, 0, stream>>>(xl_bf, xr_bf, ei1_src, ei1_dst, att1, alpha1, m1, E1V);
    edge_exp_kernel<H1V><<<(unsigned)(((size_t)E1V * H1V + 255) / 256), 256, 0, stream>>>(alpha1, m1, den1, ei1_dst, E1V);
    edge_aggr_kernel<H1V><<<E1V, H1V * 64, 0, stream>>>(xl_bf, alpha1, den1, ei1_src, ei1_dst, hacc, E1V);

    // ---- batchnorm + relu ----
    bn_stats_kernel<<<(N2V + 63) / 64, D1V, 0, stream>>>(hacc, bias1, bsum, bsq, N2V);
    bn_apply_kernel<<<N2V, D1V, 0, stream>>>(hacc, bias1, bsum, bsq, gamma, beta, hbn_bf, N2V);

    // ---- layer 2 projections (bf16 WMMA) ----
    wmma_gemm_bias_kernel<<<gemm_blocks(N2V, OUTV), 256, 0, stream>>>(hbn_bf, W2lt, b2l, yl_bf, N2V, OUTV, D1V);
    wmma_gemm_bias_kernel<<<gemm_blocks(N3V, OUTV), 256, 0, stream>>>(hbn_bf, W2rt, b2r, yr_bf, N3V, OUTV, D1V);

    // ---- layer 2 segment softmax + aggregation ----
    fill(m2, NEG_INF, (size_t)N3V * H2V);
    fill(den2, 0.f, (size_t)N3V * H2V);
    fill(out2, 0.f, (size_t)N3V * OUTV);
    edge_alpha_kernel<H2V><<<(E2V + 7) / 8, 256, 0, stream>>>(yl_bf, yr_bf, ei2_src, ei2_dst, att2, alpha2, m2, E2V);
    edge_exp_kernel<H2V><<<(unsigned)(((size_t)E2V * H2V + 255) / 256), 256, 0, stream>>>(alpha2, m2, den2, ei2_dst, E2V);
    edge_aggr_kernel<H2V><<<E2V, H2V * 64, 0, stream>>>(yl_bf, alpha2, den2, ei2_src, ei2_dst, out2, E2V);

    // ---- bias + log_softmax ----
    logsoftmax_kernel<<<(unsigned)(((size_t)N3V * 32 + 255) / 256), 256, 0, stream>>>(out2, bias2, out, N3V);
}
